// PNAModel_48275432407305
// MI455X (gfx1250) — matrix operand order repaired and checked by reading, not compiled, not run on hardware
//
#include <hip/hip_runtime.h>
#include <hip/hip_bf16.h>
#include <math.h>

#define N_NODES 20000
#define N_EDGES 40000
#define N_GRAPHS_ 512
#define TOWERS 5
#define F_IN 75
#define F_OUT 15
#define N_LAYERS 4
#define EPSV 1e-5f

typedef __attribute__((ext_vector_type(16))) _Float16 v16h;
typedef __attribute__((ext_vector_type(8)))  float    v8f;

// ---------------- WMMA helpers (CDNA5 wave32, 16x16x32 f16 -> f32) ----------

static __device__ __forceinline__ v8f wmma32(v16h a, v16h b, v8f c) {
  // D = A*B + C ; emits v_wmma_f32_16x16x32_f16
  return __builtin_amdgcn_wmma_f32_16x16x32_f16(false, a, false, b, (short)0, c,
                                                false, false);
}

// A-fragment (16x32, f16): lane holds row (lane&15); halves are two contiguous
// 8-half chunks at K = kt*32 + 8*grp and K = kt*32 + 16 + 8*grp.
static __device__ __forceinline__ v16h load_afrag(const _Float16* rowbase,
                                                  int ktile, int grp) {
  const _Float16* p = rowbase + ktile * 32 + grp * 8;
  v16h a;
#pragma unroll
  for (int i = 0; i < 8; ++i) a[i] = p[i];
#pragma unroll
  for (int i = 0; i < 8; ++i) a[8 + i] = p[16 + i];
  return a;
}

// order-preserving float <-> uint encoding for atomicMin/Max
static __device__ __forceinline__ unsigned fenc(float f) {
  unsigned u = __float_as_uint(f);
  return (u >> 31) ? ~u : (u | 0x80000000u);
}
static __device__ __forceinline__ float fdec(unsigned u) {
  u = (u >> 31) ? (u & 0x7FFFFFFFu) : ~u;
  return __uint_as_float(u);
}

// ---------------- weight conversion into B-fragment order -------------------
// dst layout: [mat][ktile][ntile][lane(32)][half(16)], B element (k,n):
//   k = kt*32 + 8*(lane>>4) + h (h<8), kt*32 + 16 + 8*(lane>>4) + (h-8) (h>=8) ;  n = nt*16 + (lane&15)
__global__ void k_conv_wB(_Float16* dst, const float* src, int nmat, int Ksrc,
                          int Nsrc, int ktiles, int ntiles) {
  long total = (long)nmat * ktiles * ntiles * 512;
  for (long idx = (long)blockIdx.x * blockDim.x + threadIdx.x; idx < total;
       idx += (long)gridDim.x * blockDim.x) {
    int q = (int)(idx % 512);
    long r0 = idx / 512;
    int nt = (int)(r0 % ntiles);
    long r1 = r0 / ntiles;
    int kt = (int)(r1 % ktiles);
    int mat = (int)(r1 / ktiles);
    int lane = q / 16, hh = q % 16;
    int grp = lane >> 4, nn = lane & 15;
    int k = kt * 32 + (hh < 8 ? grp * 8 + hh : 16 + grp * 8 + (hh - 8));
    int n = nt * 16 + nn;
    float v = (k < Ksrc && n < Nsrc) ? src[((long)mat * Ksrc + k) * Nsrc + n] : 0.f;
    dst[idx] = (_Float16)v;
  }
}

// ---------------- misc small kernels ----------------------------------------
__global__ void k_fill_u32(unsigned* p, unsigned v, long n) {
  for (long i = (long)blockIdx.x * blockDim.x + threadIdx.x; i < n;
       i += (long)gridDim.x * blockDim.x) p[i] = v;
}

__global__ void k_init_agg(float* s, float* q, unsigned* mn, unsigned* mx, long n) {
  for (long i = (long)blockIdx.x * blockDim.x + threadIdx.x; i < n;
       i += (long)gridDim.x * blockDim.x) {
    s[i] = 0.f; q[i] = 0.f; mn[i] = 0xFFFFFFFFu; mx[i] = 0u;
  }
}

__global__ void k_embed_nodes(const int* x, const float* node_emb, float* h32,
                              _Float16* h16) {
  long n = (long)N_NODES * 80;
  for (long i = (long)blockIdx.x * blockDim.x + threadIdx.x; i < n;
       i += (long)gridDim.x * blockDim.x) {
    int c = (int)(i % 80);
    long node = i / 80;
    if (c < F_IN) {
      float v = node_emb[(long)x[node] * F_IN + c];
      h32[node * F_IN + c] = v;
      h16[i] = (_Float16)v;
    } else {
      h16[i] = (_Float16)0.f;
    }
  }
}

__global__ void k_embed_edges(const int* ea, const float* edge_emb, _Float16* e50) {
  long n = (long)N_EDGES * 64;
  for (long i = (long)blockIdx.x * blockDim.x + threadIdx.x; i < n;
       i += (long)gridDim.x * blockDim.x) {
    int c = (int)(i % 64);
    long e = i / 64;
    e50[i] = (c < 50) ? (_Float16)edge_emb[(long)ea[e] * 50 + c] : (_Float16)0.f;
  }
}

__global__ void k_deg(const int* ei, float* deg) {
  int e = blockIdx.x * blockDim.x + threadIdx.x;
  if (e < N_EDGES) atomicAdd(&deg[ei[N_EDGES + e]], 1.f);
}

__global__ void k_avglog(const int* degi, float* out) {
  __shared__ float s1[32], s2[32];
  int t = threadIdx.x;  // 32 threads
  float d = (float)degi[t];
  s1[t] = logf((float)t + 1.f) * d;
  s2[t] = d;
  __syncthreads();
  if (t == 0) {
    float a = 0.f, b = 0.f;
    for (int i = 0; i < 32; ++i) { a += s1[i]; b += s2[i]; }
    out[0] = a / fmaxf(b, 1.f);
  }
}

// ---------------- edge encoder GEMM: ee = e50 @ encW + enc_b ----------------
// M tiles over edges (E/16 blocks), K = 64 (2 ktiles), N = 80 (5 ntiles)
__global__ void k_enc(const _Float16* e50, const _Float16* wB,
                      const float* enc_b, _Float16* ee) {
  int tile = blockIdx.x;              // blockDim = 64 (2 waves)
  int lane = threadIdx.x & 31, wave = threadIdx.x >> 5;
  int grp = lane >> 4;
  const _Float16* rowbase = e50 + (long)(tile * 16 + (lane & 15)) * 64;
  for (int nt = wave; nt < 5; nt += 2) {
    v8f acc = {};
#pragma unroll
    for (int kt = 0; kt < 2; ++kt) {
      v16h a = load_afrag(rowbase, kt, grp);
      v16h b = *(const v16h*)(wB + (long)(kt * 5 + nt) * 512 + lane * 16);
      acc = wmma32(a, b, acc);
    }
    int n = nt * 16 + (lane & 15);
    float bias = (n < F_IN) ? enc_b[n] : 0.f;
#pragma unroll
    for (int r = 0; r < 8; ++r) {
      int e = tile * 16 + r + 8 * grp;
      float v = (n < F_IN) ? (acc[r] + bias) : 0.f;
      ee[(long)e * 80 + n] = (_Float16)v;
    }
  }
}

// ---------------- pre-MLP GEMM + segment aggregation ------------------------
// A row (edge) = [h[dst](75) | h[src](75) | ee(75) | pad] K=256, 5 towers x 5 ntiles
__global__ void k_pre(const _Float16* h16, const _Float16* ee, const int* ei,
                      const _Float16* wB, const float* pb, float* sum, float* sq,
                      unsigned* mn, unsigned* mx) {
  __shared__ _Float16 A[16 * 256];
  __shared__ int dsts[16];
  int tile = blockIdx.x;  // blockDim = 128 (4 waves)
  int tid = threadIdx.x;
  if (tid < 16) dsts[tid] = ei[N_EDGES + tile * 16 + tid];
  for (int idx = tid; idx < 16 * 256; idx += 128) {
    int row = idx >> 8, col = idx & 255;
    int e = tile * 16 + row;
    _Float16 v = (_Float16)0.f;
    if (col < 75)       v = h16[(long)ei[N_EDGES + e] * 80 + col];
    else if (col < 150) v = h16[(long)ei[e] * 80 + (col - 75)];
    else if (col < 225) v = ee[(long)e * 80 + (col - 150)];
    A[idx] = v;
  }
  __syncthreads();
  int lane = tid & 31, wave = tid >> 5;
  int grp = lane >> 4;
  const _Float16* rowbase = A + (lane & 15) * 256;
  for (int job = wave; job < 25; job += 4) {
    int t = job / 5, nt = job % 5;
    v8f acc = {};
#pragma unroll
    for (int kt = 0; kt < 8; ++kt) {
      v16h a = load_afrag(rowbase, kt, grp);
      v16h b = *(const v16h*)(wB + (long)((t * 8 + kt) * 5 + nt) * 512 + lane * 16);
      acc = wmma32(a, b, acc);
    }
    int n = nt * 16 + (lane & 15);
    if (n < F_IN) {
      float bias = pb[t * F_IN + n];
#pragma unroll
      for (int r = 0; r < 8; ++r) {
        int row = r + 8 * grp;
        float v = acc[r] + bias;
        long off = ((long)dsts[row] * TOWERS + t) * F_IN + n;
        atomicAdd(&sum[off], v);
        atomicAdd(&sq[off], v * v);
        unsigned enc = fenc(v);
        atomicMax(&mx[off], enc);
        atomicMin(&mn[off], enc);
      }
    }
  }
}

// ---------------- post-MLP GEMM ---------------------------------------------
// A row (node,tower) = [h(75) | agg(300) | agg*u(300) | agg*d(300) | pad] K=992
__global__ void k_post(const _Float16* h16, const float* sum, const float* sq,
                       const unsigned* mn, const unsigned* mx, const float* deg,
                       const float* avg_log_p, const _Float16* wB,
                       const float* pob, _Float16* post16) {
  __shared__ _Float16 A[16 * 992];
  __shared__ float pacc[4][32][8];
  int tile = blockIdx.x;  // blockDim = 128 (4 waves)
  int tid = threadIdx.x;
  float avg_log = avg_log_p[0];
  // zero the K-pad columns (75..95) of post16 for this node tile
  for (int i = tid; i < 16 * 21; i += 128) {
    int row = i / 21, c = 75 + i % 21;
    post16[(long)(tile * 16 + row) * 96 + c] = (_Float16)0.f;
  }
  int lane = tid & 31, wave = tid >> 5;
  int grp = lane >> 4;
  const _Float16* rowbase = A + (lane & 15) * 992;
  for (int t = 0; t < TOWERS; ++t) {
    __syncthreads();
    for (int idx = tid; idx < 16 * 992; idx += 128) {
      int row = idx / 992, col = idx % 992;
      int node = tile * 16 + row;
      float v = 0.f;
      if (col < 75) {
        v = (float)h16[(long)node * 80 + col];
      } else if (col < 975) {
        int j = col - 75;
        int blk = j / 300, jj = j % 300;
        int which = jj / 75, f = jj % 75;
        long off = ((long)node * TOWERS + t) * F_IN + f;
        float dn = deg[node];
        float c = fmaxf(dn, 1.f);
        float val;
        if (which == 0) {
          val = sum[off] / c;
        } else if (which == 3) {
          float me = sum[off] / c;
          float ms = sq[off] / c;
          val = sqrtf(fmaxf(ms - me * me, 0.f) + 1e-5f);
        } else if (which == 1) {
          val = (dn > 0.f) ? fdec(mn[off]) : 0.f;
        } else {
          val = (dn > 0.f) ? fdec(mx[off]) : 0.f;
        }
        if (blk == 1)      { float ld = logf(c + 1.f); val *= ld / avg_log; }
        else if (blk == 2) { float ld = logf(c + 1.f); val *= avg_log / ld; }
        v = val;
      }
      A[idx] = (_Float16)v;
    }
    __syncthreads();
    v8f acc = {};
    for (int kt = wave; kt < 31; kt += 4) {
      v16h a = load_afrag(rowbase, kt, grp);
      v16h b = *(const v16h*)(wB + (long)(t * 31 + kt) * 512 + lane * 16);
      acc = wmma32(a, b, acc);
    }
#pragma unroll
    for (int r = 0; r < 8; ++r) pacc[wave][lane][r] = acc[r];
    __syncthreads();
    if (tid < 32) {
      int n = tid & 15, g2 = tid >> 4;
      if (n < F_OUT) {
        float bias = pob[t * F_OUT + n];
#pragma unroll
        for (int r = 0; r < 8; ++r) {
          float v = pacc[0][tid][r] + pacc[1][tid][r] + pacc[2][tid][r] +
                    pacc[3][tid][r] + bias;
          int node = tile * 16 + r + 8 * g2;
          post16[(long)node * 96 + t * F_OUT + n] = (_Float16)v;
        }
      }
    }
  }
}

// ---------------- lin GEMM: hpre = post(75) @ linW(75x75) + lin_b -----------
__global__ void k_lin(const _Float16* post16, const _Float16* wB,
                      const float* lb, float* hpre) {
  int tile = blockIdx.x;  // blockDim = 64 (2 waves)
  int lane = threadIdx.x & 31, wave = threadIdx.x >> 5;
  int grp = lane >> 4;
  const _Float16* rowbase = post16 + (long)(tile * 16 + (lane & 15)) * 96;
  for (int nt = wave; nt < 5; nt += 2) {
    v8f acc = {};
#pragma unroll
    for (int kt = 0; kt < 3; ++kt) {
      v16h a = load_afrag(rowbase, kt, grp);
      v16h b = *(const v16h*)(wB + (long)(kt * 5 + nt) * 512 + lane * 16);
      acc = wmma32(a, b, acc);
    }
    int n = nt * 16 + (lane & 15);
    if (n < F_IN) {
      float bias = lb[n];
#pragma unroll
      for (int r = 0; r < 8; ++r) {
        int node = tile * 16 + r + 8 * grp;
        hpre[(long)node * F_IN + n] = acc[r] + bias;
      }
    }
  }
}

// ---------------- batchnorm -------------------------------------------------
__global__ void k_bnstats(const float* hpre, float* bnmean, float* bnvar) {
  int f = blockIdx.x;  // 75 blocks x 256 threads
  __shared__ float ss[256], qq[256];
  float s = 0.f, q = 0.f;
  for (int n = threadIdx.x; n < N_NODES; n += 256) {
    float v = hpre[(long)n * F_IN + f];
    s += v; q += v * v;
  }
  ss[threadIdx.x] = s; qq[threadIdx.x] = q;
  __syncthreads();
  for (int st = 128; st > 0; st >>= 1) {
    if ((int)threadIdx.x < st) {
      ss[threadIdx.x] += ss[threadIdx.x + st];
      qq[threadIdx.x] += qq[threadIdx.x + st];
    }
    __syncthreads();
  }
  if (threadIdx.x == 0) {
    float mu = ss[0] / (float)N_NODES;
    bnmean[f] = mu;
    bnvar[f] = qq[0] / (float)N_NODES - mu * mu;
  }
}

__global__ void k_bnapply(const float* hpre, const float* bnmean,
                          const float* bnvar, const float* g, const float* b,
                          float* h32, _Float16* h16) {
  long n = (long)N_NODES * F_IN;
  for (long i = (long)blockIdx.x * blockDim.x + threadIdx.x; i < n;
       i += (long)gridDim.x * blockDim.x) {
    int f = (int)(i % F_IN);
    long node = i / F_IN;
    float v = hpre[i];
    float r = g[f] * (v - bnmean[f]) * rsqrtf(bnvar[f] + EPSV) + b[f];
    r = fmaxf(r, 0.f);
    h32[i] = r;
    h16[node * 80 + f] = (_Float16)r;
  }
}

// ---------------- pooling + readout MLP -------------------------------------
__global__ void k_pool(const float* h32, const int* batch, float* pooled) {
  long n = (long)N_NODES * F_IN;
  for (long i = (long)blockIdx.x * blockDim.x + threadIdx.x; i < n;
       i += (long)gridDim.x * blockDim.x) {
    int f = (int)(i % F_IN);
    long node = i / F_IN;
    atomicAdd(&pooled[(long)batch[node] * F_IN + f], h32[i]);
  }
}

__global__ void k_mlp(const float* pooled, const float* W1, const float* b1,
                      const float* W2, const float* b2, const float* W3,
                      const float* b3, float* out) {
  int g = blockIdx.x * blockDim.x + threadIdx.x;
  if (g >= N_GRAPHS_) return;
  float z1[50];
  for (int j = 0; j < 50; ++j) {
    float a = b1[j];
    for (int i = 0; i < F_IN; ++i) a += pooled[(long)g * F_IN + i] * W1[i * 50 + j];
    z1[j] = fmaxf(a, 0.f);
  }
  float z2[25];
  for (int j = 0; j < 25; ++j) {
    float a = b2[j];
    for (int i = 0; i < 50; ++i) a += z1[i] * W2[i * 25 + j];
    z2[j] = fmaxf(a, 0.f);
  }
  float a = b3[0];
  for (int i = 0; i < 25; ++i) a += z2[i] * W3[i];
  out[g] = a;
}

// ---------------- host launcher ---------------------------------------------
extern "C" void kernel_launch(void* const* d_in, const int* in_sizes, int n_in,
                              void* d_out, int out_size, void* d_ws,
                              size_t ws_size, hipStream_t stream) {
  (void)in_sizes; (void)n_in; (void)out_size; (void)ws_size;
  const int*   x        = (const int*)d_in[0];
  const int*   ei       = (const int*)d_in[1];
  const int*   ea       = (const int*)d_in[2];
  const int*   batch    = (const int*)d_in[3];
  const int*   degi     = (const int*)d_in[4];
  const float* node_emb = (const float*)d_in[5];
  const float* edge_emb = (const float*)d_in[6];
  const float* pre_W    = (const float*)d_in[7];
  const float* pre_b    = (const float*)d_in[8];
  const float* enc_W    = (const float*)d_in[9];
  const float* enc_b    = (const float*)d_in[10];
  const float* post_W   = (const float*)d_in[11];
  const float* post_b   = (const float*)d_in[12];
  const float* lin_W    = (const float*)d_in[13];
  const float* lin_b    = (const float*)d_in[14];
  const float* bn_g     = (const float*)d_in[15];
  const float* bn_b     = (const float*)d_in[16];
  const float* mlp_W1   = (const float*)d_in[17];
  const float* mlp_b1   = (const float*)d_in[18];
  const float* mlp_W2   = (const float*)d_in[19];
  const float* mlp_b2   = (const float*)d_in[20];
  const float* mlp_W3   = (const float*)d_in[21];
  const float* mlp_b3   = (const float*)d_in[22];
  float* out = (float*)d_out;

  char* ws = (char*)d_ws;
  size_t off = 0;
  auto alloc = [&](size_t bytes) -> void* {
    void* p = ws + off;
    off = (off + bytes + 255) & ~(size_t)255;
    return p;
  };

  const long NT75 = (long)N_NODES * TOWERS * F_IN;  // 7.5M
  _Float16* w_pre16  = (_Float16*)alloc((size_t)N_LAYERS * TOWERS * 8 * 5 * 512 * 2);
  _Float16* w_post16 = (_Float16*)alloc((size_t)N_LAYERS * TOWERS * 31 * 1 * 512 * 2);
  _Float16* w_enc16  = (_Float16*)alloc((size_t)N_LAYERS * 2 * 5 * 512 * 2);
  _Float16* w_lin16  = (_Float16*)alloc((size_t)N_LAYERS * 3 * 5 * 512 * 2);
  _Float16* h16      = (_Float16*)alloc((size_t)N_NODES * 80 * 2);
  float*    h32      = (float*)alloc((size_t)N_NODES * F_IN * 4);
  _Float16* e50      = (_Float16*)alloc((size_t)N_EDGES * 64 * 2);
  _Float16* ee16     = (_Float16*)alloc((size_t)N_EDGES * 80 * 2);
  float*    degn     = (float*)alloc((size_t)N_NODES * 4);
  float*    avglog   = (float*)alloc(256);
  float*    aggsum   = (float*)alloc((size_t)NT75 * 4);
  float*    aggsq    = (float*)alloc((size_t)NT75 * 4);
  unsigned* aggmn    = (unsigned*)alloc((size_t)NT75 * 4);
  unsigned* aggmx    = (unsigned*)alloc((size_t)NT75 * 4);
  _Float16* post16   = (_Float16*)alloc((size_t)N_NODES * 96 * 2);
  float*    hpre     = (float*)alloc((size_t)N_NODES * F_IN * 4);
  float*    bnmean   = (float*)alloc((size_t)F_IN * 4);
  float*    bnvar    = (float*)alloc((size_t)F_IN * 4);
  float*    pooled   = (float*)alloc((size_t)N_GRAPHS_ * F_IN * 4);

  // weights -> WMMA B-fragment order (f16, zero-padded)
  k_conv_wB<<<512, 256, 0, stream>>>(w_pre16, pre_W, N_LAYERS * TOWERS, 3 * F_IN, F_IN, 8, 5);
  k_conv_wB<<<512, 256, 0, stream>>>(w_post16, post_W, N_LAYERS * TOWERS, 13 * F_IN, F_OUT, 31, 1);
  k_conv_wB<<<64, 256, 0, stream>>>(w_enc16, enc_W, N_LAYERS, 50, F_IN, 2, 5);
  k_conv_wB<<<64, 256, 0, stream>>>(w_lin16, lin_W, N_LAYERS, F_IN, F_IN, 3, 5);

  // embeddings, degree, avg_log
  k_embed_nodes<<<(N_NODES * 80 + 255) / 256, 256, 0, stream>>>(x, node_emb, h32, h16);
  k_embed_edges<<<(N_EDGES * 64 + 255) / 256, 256, 0, stream>>>(ea, edge_emb, e50);
  k_fill_u32<<<(N_NODES + 255) / 256, 256, 0, stream>>>((unsigned*)degn, 0u, N_NODES);
  k_deg<<<(N_EDGES + 255) / 256, 256, 0, stream>>>(ei, degn);
  k_avglog<<<1, 32, 0, stream>>>(degi, avglog);

  const int ETILES = N_EDGES / 16;   // 2500
  const int NTILES = N_NODES / 16;   // 1250

  for (int l = 0; l < N_LAYERS; ++l) {
    k_init_agg<<<(int)((NT75 + 255) / 256), 256, 0, stream>>>(aggsum, aggsq, aggmn, aggmx, NT75);
    k_enc<<<ETILES, 64, 0, stream>>>(e50, w_enc16 + (size_t)l * 2 * 5 * 512,
                                     enc_b + (size_t)l * F_IN, ee16);
    k_pre<<<ETILES, 128, 0, stream>>>(h16, ee16, ei,
                                      w_pre16 + (size_t)l * TOWERS * 8 * 5 * 512,
                                      pre_b + (size_t)l * TOWERS * F_IN,
                                      aggsum, aggsq, aggmn, aggmx);
    k_post<<<NTILES, 128, 0, stream>>>(h16, aggsum, aggsq, aggmn, aggmx, degn, avglog,
                                       w_post16 + (size_t)l * TOWERS * 31 * 512,
                                       post_b + (size_t)l * TOWERS * F_OUT, post16);
    k_lin<<<NTILES, 64, 0, stream>>>(post16, w_lin16 + (size_t)l * 3 * 5 * 512,
                                     lin_b + (size_t)l * F_IN, hpre);
    k_bnstats<<<F_IN, 256, 0, stream>>>(hpre, bnmean, bnvar);
    k_bnapply<<<(N_NODES * F_IN + 255) / 256, 256, 0, stream>>>(
        hpre, bnmean, bnvar, bn_g + (size_t)l * F_IN, bn_b + (size_t)l * F_IN, h32, h16);
  }

  k_fill_u32<<<(N_GRAPHS_ * F_IN + 255) / 256, 256, 0, stream>>>(
      (unsigned*)pooled, 0u, (long)N_GRAPHS_ * F_IN);
  k_pool<<<(N_NODES * F_IN + 255) / 256, 256, 0, stream>>>(h32, batch, pooled);
  k_mlp<<<(N_GRAPHS_ + 255) / 256, 256, 0, stream>>>(pooled, mlp_W1, mlp_b1, mlp_W2,
                                                     mlp_b2, mlp_W3, mlp_b3, out);
}
